// LightWeight_ConvCIFAR10_KAN_Discriminator_33921651704117
// MI455X (gfx1250) — compile-verified, matrix operand order checked by reading, not run on hardware
//
#include <hip/hip_runtime.h>

// ---------------------------------------------------------------------------
// Types matching gfx1250 builtin signatures
// ---------------------------------------------------------------------------
typedef __attribute__((ext_vector_type(16))) __bf16 v16bf;
typedef __attribute__((ext_vector_type(8)))  __bf16 v8bf;
typedef __attribute__((ext_vector_type(8)))  float  v8f;
typedef __attribute__((ext_vector_type(4)))  unsigned int v4u;
typedef __attribute__((ext_vector_type(8)))  int v8i;
typedef __attribute__((ext_vector_type(4)))  int v4i;

#if __has_builtin(__builtin_amdgcn_tensor_load_to_lds)
#define HAVE_TDM 1
#else
#define HAVE_TDM 0
#endif

// ---------------------------------------------------------------------------
// Quadratic Cox-de-Boor B-spline bases on uniform grid g_j=(j-2)*h-1, h=2/3.
// Matches the JAX reference exactly (including evaluation at padded zeros).
// ---------------------------------------------------------------------------
__device__ __forceinline__ void bspline5(float v, float out[5]) {
    const float h    = 2.0f / 3.0f;
    const float invh = 1.5f;          // 1/h
    const float i2h  = 0.75f;         // 1/(2h)
    float g[8];
#pragma unroll
    for (int j = 0; j < 8; ++j) g[j] = (float)(j - 2) * h - 1.0f;
    float b0[7];
#pragma unroll
    for (int j = 0; j < 7; ++j) b0[j] = (v >= g[j] && v < g[j + 1]) ? 1.0f : 0.0f;
    float b1[6];
#pragma unroll
    for (int j = 0; j < 6; ++j)
        b1[j] = (v - g[j]) * invh * b0[j] + (g[j + 2] - v) * invh * b0[j + 1];
#pragma unroll
    for (int j = 0; j < 5; ++j)
        out[j] = (v - g[j]) * i2h * b1[j] + (g[j + 3] - v) * i2h * b1[j + 1];
}

// ---------------------------------------------------------------------------
// TDM: DMA a (O x 96) bf16 tile (row stride Ktot elements) from global into
// LDS at lds_base, packed rows of 96.  Issued by one wave; TENSORcnt-tracked.
// D# layout per CDNA5 ISA.  6-arg builtin (clang-23).
// ---------------------------------------------------------------------------
#if HAVE_TDM
__device__ __forceinline__ void tdm_load_wtile(const __bf16* gsrc,
                                               unsigned lds_base,
                                               int O, int Ktot) {
    unsigned long long ga = (unsigned long long)(uintptr_t)gsrc;
    v4u g0;
    g0[0] = 1u;                                             // count=1 (valid user D#)
    g0[1] = lds_base;                                       // lds_addr (bytes)
    g0[2] = (unsigned)ga;                                   // global_addr[31:0]
    g0[3] = (unsigned)((ga >> 32) & 0x1FFFFFFu) | (2u << 30); // ga[56:32] | type=2
    v8i g1;
    g1[0] = (int)(1u << 16);                                // data_size=1 -> 2 bytes
    g1[1] = (int)((unsigned)(Ktot & 0xFFFF) << 16);         // tensor_dim0[15:0]
    g1[2] = (int)(((unsigned)Ktot >> 16) |
                  ((unsigned)(O & 0xFFFF) << 16));          // dim0[31:16] | dim1[15:0]
    g1[3] = (int)(96u << 16);                               // dim1[31:16]=0 | tile_dim0=96
    g1[4] = (int)(O & 0xFFFF);                              // tile_dim1=O, tile_dim2=0
    g1[5] = (int)Ktot;                                      // tensor_dim0_stride[31:0]
    g1[6] = 0;                                              // stride[47:32] | dim1_stride lo
    g1[7] = 0;                                              // dim1_stride hi
    v4i z4; z4[0] = 0; z4[1] = 0; z4[2] = 0; z4[3] = 0;     // groups 2/3 unused (2D)
    v8i z8;
#pragma unroll
    for (int i = 0; i < 8; ++i) z8[i] = 0;
    __builtin_amdgcn_tensor_load_to_lds(g0, g1, z4, z4, z8, 0);
}
#endif

// ---------------------------------------------------------------------------
// Pack [base_w | spline_w*scaler] -> bf16 Wt of shape (O, Ktot=6F), row-major.
// K ordering: k = f*6 + {0:relu, 1..5:spline bases}.
// ---------------------------------------------------------------------------
__global__ __launch_bounds__(256) void pack_w(
    const float* __restrict__ bw, const float* __restrict__ sw,
    const float* __restrict__ sc, __bf16* __restrict__ Wt, int O, int F) {
    int i = blockIdx.x * 256 + threadIdx.x;
    if (i >= O * F) return;
    int o = i / F, f = i - o * F;
    size_t Ktot = (size_t)F * 6;
    __bf16* p = Wt + (size_t)o * Ktot + (size_t)f * 6;
    p[0] = (__bf16)bw[i];
    float s = sc[i];
#pragma unroll
    for (int j = 0; j < 5; ++j) p[1 + j] = (__bf16)(sw[(size_t)i * 5 + j] * s);
}

// ---------------------------------------------------------------------------
// Fused KAN-conv layer: im2col + relu/spline expansion (to LDS, bf16) + GEMM
// via v_wmma_f32_16x16x32_bf16.  256 threads = 8 waves; wave w owns rows
// [blk*128 + 16w, +16) of the implicit (M, 6F) feature matrix and NT=O/16
// accumulator tiles.  K tiled by 96 (= 16 raw features * 6).  Weight K-slab
// is staged into LDS by the Tensor Data Mover, overlapped with featurize.
// scsh: optional per-channel {scale[C], shift[C]} folding the previous BN.
// ---------------------------------------------------------------------------
template <int NT>
__global__ __launch_bounds__(256) void kan_gemm(
    const float* __restrict__ act, int C, int H, int W,
    const float* __restrict__ scsh,
    const __bf16* __restrict__ Wt, int Ktot,
    float* __restrict__ out) {
    const int O   = NT * 16;
    const int Hp  = H >> 1, Wp = W >> 1;
    const int HWp = Hp * Wp;

    __shared__ __bf16 smA[128 * 96];       // 24 KB expanded-feature tile
    __shared__ __bf16 smB[NT * 16 * 96];   // weight K-slab (TDM destination)

    const int tid   = threadIdx.x;
    const int lane  = tid & 31;
    const int wv    = tid >> 5;
    const int mbase = blockIdx.x * 128;

    v8f acc[NT];
#pragma unroll
    for (int n = 0; n < NT; ++n)
#pragma unroll
        for (int i = 0; i < 8; ++i) acc[n][i] = 0.0f;

    const int nK = Ktot / 96;
    for (int kt = 0; kt < nK; ++kt) {
        // L2 prefetch of next K-slab of the packed weights (global_prefetch_b8).
        if (kt + 1 < nK && tid < O)
            __builtin_prefetch(Wt + (size_t)tid * Ktot + (size_t)(kt + 1) * 96, 0, 0);

        __syncthreads();   // previous compute done: smA/smB reusable

#if HAVE_TDM
        // Wave 0 kicks the weight-tile DMA; it runs during the featurize below.
        if (wv == 0)
            tdm_load_wtile(Wt + (size_t)kt * 96,
                           (unsigned)(uintptr_t)&smB[0], O, Ktot);
#endif

        // ---- featurize 128 rows x 16 raw features into LDS (bf16) ----
#pragma unroll
        for (int it = 0; it < 8; ++it) {
            int item = tid + it * 256;          // 2048 items
            int row  = item >> 4;
            int fl   = item & 15;
            int m    = mbase + row;
            int b    = m / HWp;
            int rem  = m - b * HWp;
            int oy   = rem / Wp;
            int ox   = rem - oy * Wp;
            int f    = kt * 16 + fl;            // global raw-feature index
            int c    = f >> 4;
            int pos  = f & 15;
            int iy   = oy * 2 - 1 + (pos >> 2);
            int ix   = ox * 2 - 1 + (pos & 3);
            float v  = 0.0f;
            if (iy >= 0 && iy < H && ix >= 0 && ix < W) {
                v = act[(((size_t)b * C + c) * H + iy) * W + ix];
                if (scsh) v = v * scsh[c] + scsh[C + c];   // folded BN
            }
            float s5[5];
            bspline5(v, s5);
            __bf16* p = &smA[row * 96 + fl * 6];
            p[0] = (__bf16)fmaxf(v, 0.0f);
#pragma unroll
            for (int j = 0; j < 5; ++j) p[1 + j] = (__bf16)s5[j];
        }

#if HAVE_TDM
        if (wv == 0) __builtin_amdgcn_s_wait_tensorcnt(0);  // weight DMA landed
#else
        // Fallback: cooperative global->LDS copy of the weight slab.
        for (int i = tid; i < O * 12; i += 256) {
            int row = i / 12, seg = i - row * 12;
            *(uint4*)&smB[row * 96 + seg * 8] =
                *(const uint4*)(Wt + (size_t)row * Ktot + (size_t)kt * 96 + seg * 8);
        }
#endif
        __syncthreads();   // smA written, smB DMA visible to all waves

        // ---- 3 WMMA K-steps of 32 over this 96-wide tile ----
        const int r    = lane & 15;
        const int koff = (lane >> 4) * 8;
#pragma unroll
        for (int ks = 0; ks < 3; ++ks) {
            const int k0 = ks * 32;
            // A fragment (16x32 bf16) from LDS, documented wave32 layout.
            const __bf16* ap = &smA[(wv * 16 + r) * 96 + k0 + koff];
            v8bf alo = *(const v8bf*)ap;
            v8bf ahi = *(const v8bf*)(ap + 16);
            v16bf afrag;
#pragma unroll
            for (int i = 0; i < 8; ++i) { afrag[i] = alo[i]; afrag[8 + i] = ahi[i]; }

#pragma unroll
            for (int nt = 0; nt < NT; ++nt) {
                // B fragment from the LDS-staged W^T slab, mirrored layout.
                const __bf16* bp = &smB[(nt * 16 + r) * 96 + k0 + koff];
                v8bf blo = *(const v8bf*)bp;
                v8bf bhi = *(const v8bf*)(bp + 16);
                v16bf bfrag;
#pragma unroll
                for (int i = 0; i < 8; ++i) { bfrag[i] = blo[i]; bfrag[8 + i] = bhi[i]; }
                acc[nt] = __builtin_amdgcn_wmma_f32_16x16x32_bf16(
                    false, afrag, false, bfrag, (short)0, acc[nt], false, false);
            }
        }
    }

    // ---- scatter the 16 x O tile to NCHW f32 ----
    const int col  = lane & 15;
    const int rofs = (lane >> 4) * 8;
#pragma unroll
    for (int rr = 0; rr < 8; ++rr) {
        int m   = mbase + wv * 16 + rr + rofs;
        int b   = m / HWp;
        int rem = m - b * HWp;
        int oy  = rem / Wp;
        int ox  = rem - oy * Wp;
#pragma unroll
        for (int nt = 0; nt < NT; ++nt) {
            int oc = nt * 16 + col;
            out[((size_t)b * O + oc) * HWp + oy * Wp + ox] = acc[nt][rr];
        }
    }
}

// ---------------------------------------------------------------------------
// Train-mode BN stats: per channel mean/var over (B,H,W); emit fused
// scale = gamma*rsqrt(var+eps), shift = beta - mean*scale.
// ---------------------------------------------------------------------------
__global__ __launch_bounds__(256) void bn_stats(
    const float* __restrict__ act, int C, int HW, int B,
    const float* __restrict__ gamma, const float* __restrict__ beta,
    float* __restrict__ scsh) {
    const int c = blockIdx.x;
    const int N = B * HW;
    float s = 0.0f, s2 = 0.0f;
    for (int i = threadIdx.x; i < N; i += 256) {
        int b = i / HW, p = i - b * HW;
        float v = act[((size_t)b * C + c) * HW + p];
        s += v;
        s2 += v * v;
    }
    __shared__ float sh[512];
    sh[threadIdx.x]       = s;
    sh[256 + threadIdx.x] = s2;
    __syncthreads();
    for (int st = 128; st > 0; st >>= 1) {
        if (threadIdx.x < st) {
            sh[threadIdx.x]       += sh[threadIdx.x + st];
            sh[256 + threadIdx.x] += sh[256 + threadIdx.x + st];
        }
        __syncthreads();
    }
    if (threadIdx.x == 0) {
        float mean  = sh[0] / (float)N;
        float var   = sh[256] / (float)N - mean * mean;
        float scale = gamma[c] * rsqrtf(var + 1e-5f);
        scsh[c]     = scale;
        scsh[C + c] = beta[c] - mean * scale;
    }
}

// ---------------------------------------------------------------------------
// Head: BN(fold) + avgpool(2x2) + FC(128->1) + sigmoid. One block per batch.
// ---------------------------------------------------------------------------
__global__ __launch_bounds__(128) void head(
    const float* __restrict__ act4, const float* __restrict__ scsh,
    const float* __restrict__ fcw, const float* __restrict__ fcb,
    float* __restrict__ out) {
    const int b = blockIdx.x, c = threadIdx.x;   // C = 128, HW = 4
    float s = 0.0f;
#pragma unroll
    for (int i = 0; i < 4; ++i) s += act4[((size_t)b * 128 + c) * 4 + i];
    float x = (s * 0.25f) * scsh[c] + scsh[128 + c];
    __shared__ float sh[128];
    sh[c] = x * fcw[c];
    __syncthreads();
    for (int st = 64; st > 0; st >>= 1) {
        if (c < st) sh[c] += sh[c + st];
        __syncthreads();
    }
    if (c == 0) out[b] = 1.0f / (1.0f + expf(-(sh[0] + fcb[0])));
}

// ---------------------------------------------------------------------------
// Launch
// ---------------------------------------------------------------------------
extern "C" void kernel_launch(void* const* d_in, const int* in_sizes, int n_in,
                              void* d_out, int out_size, void* d_ws, size_t ws_size,
                              hipStream_t stream) {
    const float* x   = (const float*)d_in[0];
    const float* bw1 = (const float*)d_in[1];
    const float* sw1 = (const float*)d_in[2];
    const float* sc1 = (const float*)d_in[3];
    const float* bw2 = (const float*)d_in[4];
    const float* sw2 = (const float*)d_in[5];
    const float* sc2 = (const float*)d_in[6];
    const float* bw3 = (const float*)d_in[7];
    const float* sw3 = (const float*)d_in[8];
    const float* sc3 = (const float*)d_in[9];
    const float* bw4 = (const float*)d_in[10];
    const float* sw4 = (const float*)d_in[11];
    const float* sc4 = (const float*)d_in[12];
    const float* g1  = (const float*)d_in[13];
    const float* b1  = (const float*)d_in[14];
    const float* g2  = (const float*)d_in[15];
    const float* b2  = (const float*)d_in[16];
    const float* g3  = (const float*)d_in[17];
    const float* b3  = (const float*)d_in[18];
    const float* fcw = (const float*)d_in[19];
    const float* fcb = (const float*)d_in[20];
    float* out = (float*)d_out;

    const int B = 512;

    // Workspace carving (256B aligned slots, ~35 MB total).
    char*  ws  = (char*)d_ws;
    size_t off = 0;
    auto take = [&](size_t bytes) -> void* {
        void* p = ws + off;
        off = (off + bytes + 255) & ~(size_t)255;
        return p;
    };
    __bf16* Wt1  = (__bf16*)take((size_t)288 * 32 * 2);
    __bf16* Wt2  = (__bf16*)take((size_t)3072 * 64 * 2);
    __bf16* Wt3  = (__bf16*)take((size_t)6144 * 128 * 2);
    __bf16* Wt4  = (__bf16*)take((size_t)12288 * 128 * 2);
    float*  a1   = (float*)take((size_t)B * 32 * 16 * 16 * 4);
    float*  a2   = (float*)take((size_t)B * 64 * 8 * 8 * 4);
    float*  a3   = (float*)take((size_t)B * 128 * 4 * 4 * 4);
    float*  a4   = (float*)take((size_t)B * 128 * 2 * 2 * 4);
    float*  ss1  = (float*)take((size_t)2 * 64 * 4);
    float*  ss2  = (float*)take((size_t)2 * 128 * 4);
    float*  ss3  = (float*)take((size_t)2 * 128 * 4);
    (void)ws_size; (void)in_sizes; (void)n_in; (void)out_size;

    // Pack weights to bf16 (O, 6F).
    pack_w<<<(32 * 48 + 255) / 256, 256, 0, stream>>>(bw1, sw1, sc1, Wt1, 32, 48);
    pack_w<<<(64 * 512 + 255) / 256, 256, 0, stream>>>(bw2, sw2, sc2, Wt2, 64, 512);
    pack_w<<<(128 * 1024 + 255) / 256, 256, 0, stream>>>(bw3, sw3, sc3, Wt3, 128, 1024);
    pack_w<<<(128 * 2048 + 255) / 256, 256, 0, stream>>>(bw4, sw4, sc4, Wt4, 128, 2048);

    // Layer 1: (512,3,32,32) -> (512,32,16,16); M=131072, K=288, O=32.
    kan_gemm<2><<<131072 / 128, 256, 0, stream>>>(x, 3, 32, 32, nullptr, Wt1, 288, a1);
    // Layer 2: -> (512,64,8,8); M=32768, K=3072, O=64.
    kan_gemm<4><<<32768 / 128, 256, 0, stream>>>(a1, 32, 16, 16, nullptr, Wt2, 3072, a2);
    // BN1 on a2.
    bn_stats<<<64, 256, 0, stream>>>(a2, 64, 64, B, g1, b1, ss1);
    // Layer 3 (BN1 folded): -> (512,128,4,4); M=8192, K=6144, O=128.
    kan_gemm<8><<<8192 / 128, 256, 0, stream>>>(a2, 64, 8, 8, ss1, Wt3, 6144, a3);
    // BN2 on a3.
    bn_stats<<<128, 256, 0, stream>>>(a3, 128, 16, B, g2, b2, ss2);
    // Layer 4 (BN2 folded): -> (512,128,2,2); M=2048, K=12288, O=128.
    kan_gemm<8><<<2048 / 128, 256, 0, stream>>>(a3, 128, 4, 4, ss2, Wt4, 12288, a4);
    // BN3 on a4.
    bn_stats<<<128, 256, 0, stream>>>(a4, 128, 4, B, g3, b3, ss3);
    // Head: BN3 fold + pool + FC + sigmoid.
    head<<<B, 128, 0, stream>>>(a4, ss3, fcw, fcb, out);
}
